// GraphModel_64458869179010
// MI455X (gfx1250) — compile-verified, matrix operand order checked by reading, not run on hardware
//
#include <hip/hip_runtime.h>
#include <hip/hip_fp16.h>
#include <math.h>

#define NNODES 50000
#define NEDGES 800000
#define INF_   64
#define DF     128
#define NH     4
#define NL     3
#define NC     3
#define NV     512
#define NOUT   16
#define LDH    (NL * DF)   // 384

typedef __attribute__((ext_vector_type(16))) _Float16 v16h;
typedef __attribute__((ext_vector_type(8)))  float    v8f;
typedef __attribute__((ext_vector_type(4)))  float    v4f;

// ---------- helpers ----------
__device__ __forceinline__ unsigned encodeOrd(float f) {
    unsigned u = __float_as_uint(f);
    return (u & 0x80000000u) ? ~u : (u | 0x80000000u);
}
__device__ __forceinline__ float decodeOrd(unsigned u) {
    return __uint_as_float((u & 0x80000000u) ? (u & 0x7FFFFFFFu) : ~u);
}

// ---------- init kernels ----------
__global__ void k_zero_f4(float4* p, int n4) {
    int i = blockIdx.x * blockDim.x + threadIdx.x;
    if (i < n4) p[i] = make_float4(0.f, 0.f, 0.f, 0.f);
}
__global__ void k_zero_f(float* p, int n) {
    int i = blockIdx.x * blockDim.x + threadIdx.x;
    if (i < n) p[i] = 0.f;
}
__global__ void k_init_max(unsigned* p, int n) {
    int i = blockIdx.x * blockDim.x + threadIdx.x;
    if (i < n) p[i] = 0x007FFFFFu;   // encodeOrd(-inf)
}

// ---------- weight prep: fp32 W[K,N] -> f16 swizzled WMMA-B fragments ----------
// Tiles of (32 K x 16 N); within a tile, 32 lanes x 16 halfs contiguous.
// lane = {g=lane>>4, n=lane&15}; element j -> K = (j<8 ? g*8+j : 16+g*8+(j-8)).
__global__ void k_prep_w(const float* __restrict__ W, _Float16* __restrict__ Bsw,
                         int K, int N) {
    int t = blockIdx.x * blockDim.x + threadIdx.x;
    int total = K * N;
    if (t >= total) return;
    int tile = t >> 9;          // 512 elements per tile
    int e    = t & 511;
    int lane = e >> 4;
    int j    = e & 15;
    int ntn  = N >> 4;
    int kt   = tile / ntn;
    int nt   = tile - kt * ntn;
    int g    = lane >> 4;
    int n    = nt * 16 + (lane & 15);
    int k    = kt * 32 + ((j < 8) ? (g * 8 + j) : (16 + g * 8 + (j - 8)));
    Bsw[t] = (_Float16)W[k * N + n];
}

// ---------- head fusion prep: Wc = head_w @ p_w (384x16), bc = head_b @ p_w + p_b ----------
__global__ void k_prep_head(const float* __restrict__ head_w, const float* __restrict__ head_b,
                            const float* __restrict__ p_w, const float* __restrict__ p_b,
                            float* __restrict__ Wc, float* __restrict__ bc) {
    int t = blockIdx.x * blockDim.x + threadIdx.x;
    if (t < LDH * NOUT) {
        int k = t / NOUT, o = t - k * NOUT;
        float s = 0.f;
        for (int v = 0; v < NV; ++v) s += head_w[k * NV + v] * p_w[v * NOUT + o];
        Wc[t] = s;
    } else if (t < LDH * NOUT + NOUT) {
        int o = t - LDH * NOUT;
        float s = p_b[o];
        for (int v = 0; v < NV; ++v) s += head_b[v] * p_w[v * NOUT + o];
        bc[o] = s;
    }
}

// ---------- feature GEMM: feat[M,128] = A[M,K](fp32, ld=lda) x W(K,128) via f16 WMMA ----------
// 256 threads = 8 waves as 2(M) x 4(N) grid of 16x16 tiles -> block tile 32x64.
// A-row index is CLAMPED (not predicated): an A row only feeds the same-numbered
// D row, and out-of-range D rows are never stored, so the hot loop stays
// branch-free with EXEC all-1s (required for WMMA, avoids saveexec per step).
__global__ void k_gemm_wmma(const float* __restrict__ A, int lda,
                            const _Float16* __restrict__ Bsw,
                            float* __restrict__ Cd, int ldc,
                            int M, int K, int Ncols) {
    const int lane = threadIdx.x & 31;
    const int wave = threadIdx.x >> 5;
    const int wm   = wave >> 2;
    const int wn   = wave & 3;
    const int rb   = blockIdx.x * 32 + wm * 16;
    const int g    = lane >> 4;
    const int l16  = lane & 15;
    int m          = rb + l16;
    m = (m < M) ? m : (M - 1);                 // clamp: keeps loads valid, no divergence
    const int ntn  = Ncols >> 4;
    const int ntile = blockIdx.y * 4 + wn;
    const int cb   = ntile * 16;

    const float*    ap = A + (size_t)m * lda + g * 8;
    const _Float16* bp = Bsw + ((size_t)ntile << 9) + (lane << 4);
    const size_t bstep = (size_t)ntn << 9;

    v8f acc = {};
    for (int k0 = 0; k0 < K; k0 += 32) {
        v4f a0 = *(const v4f*)(ap);
        v4f a1 = *(const v4f*)(ap + 4);
        v4f a2 = *(const v4f*)(ap + 16);
        v4f a3 = *(const v4f*)(ap + 20);
        v16h a;
#pragma unroll
        for (int j = 0; j < 4; ++j) {
            a[j]      = (_Float16)a0[j];
            a[4 + j]  = (_Float16)a1[j];
            a[8 + j]  = (_Float16)a2[j];
            a[12 + j] = (_Float16)a3[j];
        }
        v16h b = *(const v16h*)bp;
        acc = __builtin_amdgcn_wmma_f32_16x16x32_f16(false, a, false, b,
                                                     (short)0, acc, false, false);
        ap += 32;
        bp += bstep;
    }
#pragma unroll
    for (int r = 0; r < 8; ++r) {
        int row = rb + g * 8 + r;
        if (row < M) Cd[(size_t)row * ldc + cb + l16] = acc[r];
    }
}

// ---------- edge pass 1: coord-attention logits + per-dst atomic max ----------
__global__ void k_edge_logits(const float* __restrict__ coords,
                              const int* __restrict__ src, const int* __restrict__ dst,
                              const float* __restrict__ Al,   // [C,H] for this layer
                              const float* __restrict__ bAl,  // [H]
                              float* __restrict__ logitBuf,   // [E,H]
                              unsigned* __restrict__ nodeMax) // [N,H]
{
    int e = blockIdx.x * blockDim.x + threadIdx.x;
    if (e >= NEDGES) return;
    int s = src[e], d = dst[e];
    float r0 = coords[d * 3 + 0] - coords[s * 3 + 0];
    float r1 = coords[d * 3 + 1] - coords[s * 3 + 1];
    float r2 = coords[d * 3 + 2] - coords[s * 3 + 2];
#pragma unroll
    for (int h = 0; h < NH; ++h) {
        float z = bAl[h] + r0 * Al[0 * NH + h] + r1 * Al[1 * NH + h] + r2 * Al[2 * NH + h];
        z = (z > 0.f) ? z : 0.2f * z;                 // leaky_relu(0.2)
        logitBuf[(size_t)e * NH + h] = z;
        atomicMax(&nodeMax[(size_t)d * NH + h], encodeOrd(z));
    }
}

// ---------- edge pass 2: exp(logit - max) + per-dst atomic sum ----------
__global__ void k_edge_expsum(const int* __restrict__ dst,
                              float* __restrict__ logitBuf,
                              const unsigned* __restrict__ nodeMax,
                              float* __restrict__ nodeSum) {
    int e = blockIdx.x * blockDim.x + threadIdx.x;
    if (e >= NEDGES) return;
    int d = dst[e];
#pragma unroll
    for (int h = 0; h < NH; ++h) {
        float m  = decodeOrd(nodeMax[(size_t)d * NH + h]);
        float ex = expf(logitBuf[(size_t)e * NH + h] - m);
        logitBuf[(size_t)e * NH + h] = ex;
        atomicAdd(&nodeSum[(size_t)d * NH + h], ex);
    }
}

// ---------- edge pass 3: weighted message scatter. One wave per edge, 4 floats/lane ----------
__global__ void k_scatter(const int* __restrict__ src, const int* __restrict__ dst,
                          const float* __restrict__ feat,      // [N,128]
                          const float* __restrict__ eBuf,      // exp values [E,H]
                          const float* __restrict__ nodeSum,   // [N,H]
                          float* __restrict__ aggBase)         // gh + l*DF, row stride LDH
{
    int e = blockIdx.x * 8 + (threadIdx.x >> 5);
    if (e >= NEDGES) return;
    int lane = threadIdx.x & 31;
    int s = src[e], d = dst[e];
    int head = lane >> 3;                             // 32 floats per head, 4 per lane
    float alpha = eBuf[(size_t)e * NH + head] /
                  (nodeSum[(size_t)d * NH + head] + 1e-9f);
    v4f f = *(const v4f*)(feat + (size_t)s * DF + lane * 4);
    float* outp = aggBase + (size_t)d * LDH + lane * 4;
    atomicAdd(outp + 0, f.x * alpha);
    atomicAdd(outp + 1, f.y * alpha);
    atomicAdd(outp + 2, f.z * alpha);
    atomicAdd(outp + 3, f.w * alpha);
}

// ---------- leaky_relu(0.01) in place on a gh slice ----------
__global__ void k_leaky01(float* __restrict__ slice) {   // row stride LDH, DF cols
    int i = blockIdx.x * blockDim.x + threadIdx.x;
    if (i >= NNODES * DF) return;
    int row = i / DF, col = i - row * DF;
    float* p = slice + (size_t)row * LDH + col;
    float v = *p;
    *p = (v > 0.f) ? v : 0.01f * v;
}

// ---------- final: LayerNorm(384) + fused head (x Wc[384x16] + bc). One wave per node ----------
__global__ void k_final(const float* __restrict__ gh,
                        const float* __restrict__ gamma, const float* __restrict__ beta,
                        const float* __restrict__ Wc, const float* __restrict__ bc,
                        float* __restrict__ out) {
    int node = blockIdx.x * 8 + (threadIdx.x >> 5);
    if (node >= NNODES) return;
    int lane = threadIdx.x & 31;
    const float* row = gh + (size_t)node * LDH;
    float vals[12];
    float s = 0.f, sq = 0.f;
#pragma unroll
    for (int j = 0; j < 12; ++j) {
        float v = row[lane + 32 * j];
        vals[j] = v; s += v; sq += v * v;
    }
    for (int off = 16; off > 0; off >>= 1) {
        s  += __shfl_xor(s, off);
        sq += __shfl_xor(sq, off);
    }
    float mu   = s * (1.f / LDH);
    float var  = sq * (1.f / LDH) - mu * mu;
    float rstd = rsqrtf(var + 1e-5f);
    float acc[NOUT];
#pragma unroll
    for (int o = 0; o < NOUT; ++o) acc[o] = 0.f;
#pragma unroll
    for (int j = 0; j < 12; ++j) {
        int k = lane + 32 * j;
        float hn = (vals[j] - mu) * rstd * gamma[k] + beta[k];
        const float* wrow = Wc + (size_t)k * NOUT;
#pragma unroll
        for (int o = 0; o < NOUT; ++o) acc[o] += hn * wrow[o];
    }
    for (int off = 16; off > 0; off >>= 1) {
#pragma unroll
        for (int o = 0; o < NOUT; ++o) acc[o] += __shfl_xor(acc[o], off);
    }
    if (lane == 0) {
#pragma unroll
        for (int o = 0; o < NOUT; ++o) out[(size_t)node * NOUT + o] = acc[o] + bc[o];
    }
}

// ---------- launcher ----------
extern "C" void kernel_launch(void* const* d_in, const int* in_sizes, int n_in,
                              void* d_out, int out_size, void* d_ws, size_t ws_size,
                              hipStream_t stream) {
    const float* x        = (const float*)d_in[0];
    const float* coords   = (const float*)d_in[1];
    const int*   edge_src = (const int*)d_in[2];
    const int*   edge_dst = (const int*)d_in[3];
    const float* W0       = (const float*)d_in[4];
    const float* W_rest   = (const float*)d_in[5];
    const float* Aw       = (const float*)d_in[6];
    const float* bA       = (const float*)d_in[7];
    const float* ln_gamma = (const float*)d_in[8];
    const float* ln_beta  = (const float*)d_in[9];
    const float* head_w   = (const float*)d_in[10];
    const float* head_b   = (const float*)d_in[11];
    const float* p_w      = (const float*)d_in[12];
    const float* p_b      = (const float*)d_in[13];
    float* out = (float*)d_out;

    char* ws = (char*)d_ws;
    size_t off = 0;
    auto alloc = [&](size_t bytes) { char* p = ws + off; off = (off + bytes + 255) & ~(size_t)255; return p; };
    float*    feat     = (float*)alloc((size_t)NNODES * DF * 4);
    float*    gh       = (float*)alloc((size_t)NNODES * LDH * 4);
    float*    logitBuf = (float*)alloc((size_t)NEDGES * NH * 4);
    unsigned* nodeMax  = (unsigned*)alloc((size_t)NNODES * NH * 4);
    float*    nodeSum  = (float*)alloc((size_t)NNODES * NH * 4);
    float*    Wc       = (float*)alloc((size_t)LDH * NOUT * 4);
    float*    bc       = (float*)alloc((size_t)NOUT * 4);
    _Float16* Bsw0     = (_Float16*)alloc((size_t)INF_ * DF * 2);
    _Float16* Bsw1     = (_Float16*)alloc((size_t)DF * DF * 2);
    _Float16* Bsw2     = (_Float16*)alloc((size_t)DF * DF * 2);

    const int T = 256;
    // prep: zero gh, build fused head weights, swizzle layer weights to f16 fragments
    k_zero_f4<<<(NNODES * LDH / 4 + T - 1) / T, T, 0, stream>>>((float4*)gh, NNODES * LDH / 4);
    k_prep_head<<<(LDH * NOUT + NOUT + T - 1) / T, T, 0, stream>>>(head_w, head_b, p_w, p_b, Wc, bc);
    k_prep_w<<<(INF_ * DF + T - 1) / T, T, 0, stream>>>(W0, Bsw0, INF_, DF);
    k_prep_w<<<(DF * DF + T - 1) / T, T, 0, stream>>>(W_rest, Bsw1, DF, DF);
    k_prep_w<<<(DF * DF + T - 1) / T, T, 0, stream>>>(W_rest + DF * DF, Bsw2, DF, DF);

    const int egrid = (NEDGES + T - 1) / T;
    const _Float16* Bs[NL] = {Bsw0, Bsw1, Bsw2};

    for (int l = 0; l < NL; ++l) {
        // per-layer softmax accumulators
        k_init_max<<<(NNODES * NH + T - 1) / T, T, 0, stream>>>(nodeMax, NNODES * NH);
        k_zero_f<<<(NNODES * NH + T - 1) / T, T, 0, stream>>>(nodeSum, NNODES * NH);

        // feat = h @ W[l]   (WMMA f16->f32)
        const float* Ain = (l == 0) ? x : (gh + (size_t)(l - 1) * DF);
        int lda = (l == 0) ? INF_ : LDH;
        int K   = (l == 0) ? INF_ : DF;
        dim3 ggrid((NNODES + 31) / 32, DF / 64);
        k_gemm_wmma<<<ggrid, T, 0, stream>>>(Ain, lda, Bs[l], feat, DF, NNODES, K, DF);

        // attention softmax over incoming edges
        k_edge_logits<<<egrid, T, 0, stream>>>(coords, edge_src, edge_dst,
                                               Aw + l * NC * NH, bA + l * NH,
                                               logitBuf, nodeMax);
        k_edge_expsum<<<egrid, T, 0, stream>>>(edge_dst, logitBuf, nodeMax, nodeSum);

        // weighted scatter into gh slice l (one wave per edge)
        k_scatter<<<(NEDGES + 7) / 8, T, 0, stream>>>(edge_src, edge_dst, feat,
                                                      logitBuf, nodeSum,
                                                      gh + (size_t)l * DF);
        if (l != NL - 1)
            k_leaky01<<<(NNODES * DF + T - 1) / T, T, 0, stream>>>(gh + (size_t)l * DF);
    }

    // LayerNorm + fused projection head
    k_final<<<(NNODES + 7) / 8, T, 0, stream>>>(gh, ln_gamma, ln_beta, Wc, bc, out);
}
